// FrequencyGuidedAttention_9904194585051
// MI455X (gfx1250) — matrix operand order repaired and checked by reading, not compiled, not run on hardware
//
#include <hip/hip_runtime.h>
#include <hip/hip_bf16.h>

typedef __attribute__((ext_vector_type(16))) _Float16 v16h;
typedef __attribute__((ext_vector_type(8)))  _Float16 v8h;
typedef __attribute__((ext_vector_type(8)))  float    v8f;

#define HW   16384
#define IMG  128
#define NB   4
#define CDIM 384
#define NH   8
#define HC   48
#define EPSV 1e-12f

// ---------------------------------------------------------------------------
// Generic batched GEMM:  Y[o,n] = sum_i W[o,i] * X[i,n]   (f16 WMMA, f32 acc)
// Tile: 64(M) x 64(N), K-step 32. 128 threads = 4 waves; wave w owns rows
// [m0+16w, m0+16w+16). X tile staged transposed into LDS as f16 so the WMMA
// B fragment is a contiguous 32B LDS read per lane.
// Requires N % 64 == 0 (true for all call sites: N = 16384).
// ---------------------------------------------------------------------------
__launch_bounds__(128)
__global__ void gemm_wmma_kernel(const float* __restrict__ Wm,
                                 const float* __restrict__ X,
                                 float* __restrict__ Y,
                                 int O, int I, int N,
                                 long long wStride, long long xStride,
                                 long long yStride)
{
    __shared__ _Float16 Bs[64][48] __attribute__((aligned(32))); // 96B rows: 32B-aligned K-halves
    const float* Wb = Wm + wStride * blockIdx.z;
    const float* Xb = X  + xStride * blockIdx.z;
    float*       Yb = Y  + yStride * blockIdx.z;

    const int n0   = blockIdx.x * 64;
    const int m0   = blockIdx.y * 64;
    const int lane = threadIdx.x & 31;
    const int wave = threadIdx.x >> 5;
    const int kh   = lane >> 4;     // K-half selector (contiguous K order)
    const int l16  = lane & 15;
    const int arow = m0 + wave * 16 + l16;   // A-matrix row for this lane

    // staging role: this thread owns LDS column nn, K-strip [rg*16, rg*16+16)
    const int nn = threadIdx.x & 63;
    const int rg = threadIdx.x >> 6;         // 0 or 1
    const long long gcol = n0 + nn;

    v8f acc0 = {}, acc1 = {}, acc2 = {}, acc3 = {};

    for (int k0 = 0; k0 < I; k0 += 32) {
        // ---- stage X[k0:k0+32, n0:n0+64] transposed -> Bs[n][k] (f16) ----
        {
            v16h pack;
            const int kbase = k0 + rg * 16;
            if (k0 + 32 <= I) {
                // fast path: unguarded; each of the 16 loads is fully
                // coalesced across the 64 threads of this row-group
                #pragma unroll
                for (int j = 0; j < 16; ++j)
                    pack[j] = (_Float16)Xb[(long long)(kbase + j) * N + gcol];
            } else {
                #pragma unroll
                for (int j = 0; j < 16; ++j) {
                    int gi = kbase + j;
                    pack[j] = (gi < I) ? (_Float16)Xb[(long long)gi * N + gcol]
                                       : (_Float16)0.0f;
                }
            }
            *(v16h*)&Bs[nn][rg * 16] = pack;   // one 32B strip: ds_store_b128 x2
        }
        if (k0 + 32 < I)  // hint next K-tile into cache (global_prefetch_b8)
            __builtin_prefetch(&Xb[(long long)(k0 + 32 + rg * 16) * N + gcol], 0, 1);
        __syncthreads();

        // ---- A fragment from W (contiguous K order, same as B) ----
        // lanes 0-15: K = {kb..kb+15} with kb=k0 ; lanes 16-31: kb=k0+16
        v16h a;
        const int kb = k0 + kh * 16;
        if (arow < O && k0 + 32 <= I) {
            const float* wr = &Wb[(long long)arow * I + kb];
            float4 f0 = *(const float4*)(wr);
            float4 f1 = *(const float4*)(wr + 4);
            float4 f2 = *(const float4*)(wr + 8);
            float4 f3 = *(const float4*)(wr + 12);
            a[0]  = (_Float16)f0.x; a[1]  = (_Float16)f0.y;
            a[2]  = (_Float16)f0.z; a[3]  = (_Float16)f0.w;
            a[4]  = (_Float16)f1.x; a[5]  = (_Float16)f1.y;
            a[6]  = (_Float16)f1.z; a[7]  = (_Float16)f1.w;
            a[8]  = (_Float16)f2.x; a[9]  = (_Float16)f2.y;
            a[10] = (_Float16)f2.z; a[11] = (_Float16)f2.w;
            a[12] = (_Float16)f3.x; a[13] = (_Float16)f3.y;
            a[14] = (_Float16)f3.z; a[15] = (_Float16)f3.w;
        } else {
            #pragma unroll
            for (int t = 0; t < 8; ++t) {
                int kk = kb + t;
                a[t] = (arow < O && kk < I) ? (_Float16)Wb[(long long)arow * I + kk]
                                            : (_Float16)0.0f;
            }
            #pragma unroll
            for (int t = 0; t < 8; ++t) {
                int kk = kb + 8 + t;
                a[8 + t] = (arow < O && kk < I) ? (_Float16)Wb[(long long)arow * I + kk]
                                                : (_Float16)0.0f;
            }
        }

        // ---- B fragments: lane = col, 16 contiguous K halves (32B aligned) ----
        v16h b0 = *(const v16h*)&Bs[ 0 + l16][kh * 16];
        v16h b1 = *(const v16h*)&Bs[16 + l16][kh * 16];
        v16h b2 = *(const v16h*)&Bs[32 + l16][kh * 16];
        v16h b3 = *(const v16h*)&Bs[48 + l16][kh * 16];

        acc0 = __builtin_amdgcn_wmma_f32_16x16x32_f16(false, a, false, b0, (short)0, acc0, false, false);
        acc1 = __builtin_amdgcn_wmma_f32_16x16x32_f16(false, a, false, b1, (short)0, acc1, false, false);
        acc2 = __builtin_amdgcn_wmma_f32_16x16x32_f16(false, a, false, b2, (short)0, acc2, false, false);
        acc3 = __builtin_amdgcn_wmma_f32_16x16x32_f16(false, a, false, b3, (short)0, acc3, false, false);
        __syncthreads();
    }

    // ---- store: VGPR r -> M = 16*wave + 8*kh + r ; N = tile + l16 ----
    #pragma unroll
    for (int r = 0; r < 8; ++r) {
        int mm = m0 + wave * 16 + kh * 8 + r;
        if (mm < O) {
            long long base = (long long)mm * N + n0 + l16;
            Yb[base     ] = acc0[r];
            Yb[base + 16] = acc1[r];
            Yb[base + 32] = acc2[r];
            Yb[base + 48] = acc3[r];
        }
    }
}

// ---------------------------------------------------------------------------
// Depthwise 3x3, SAME zero padding (cross-correlation, matches lax/torch).
// Interior pixels take an unguarded fast path.
// ---------------------------------------------------------------------------
__global__ void dwconv3x3_kernel(const float* __restrict__ in,
                                 const float* __restrict__ w,
                                 float* __restrict__ out,
                                 int C, long long inBatchStride,
                                 long long inChanBase, int wChanOff, int total)
{
    int idx = blockIdx.x * blockDim.x + threadIdx.x;
    if (idx >= total) return;
    int x = idx & (IMG - 1);
    int y = (idx >> 7) & (IMG - 1);
    int c = (idx >> 14) % C;
    int b = idx / (C * HW);
    const float* ip = in + (long long)b * inBatchStride + inChanBase + (long long)c * HW;
    const float* wp = w + (long long)(wChanOff + c) * 9;
    float s;
    if (x >= 1 && x <= IMG - 2 && y >= 1 && y <= IMG - 2) {
        const float* p0 = ip + (y - 1) * IMG + (x - 1);
        const float* p1 = p0 + IMG;
        const float* p2 = p1 + IMG;
        s = p0[0] * wp[0] + p0[1] * wp[1] + p0[2] * wp[2]
          + p1[0] * wp[3] + p1[1] * wp[4] + p1[2] * wp[5]
          + p2[0] * wp[6] + p2[1] * wp[7] + p2[2] * wp[8];
    } else {
        s = 0.0f;
        #pragma unroll
        for (int ky = 0; ky < 3; ++ky) {
            int yy = y + ky - 1;
            if (yy < 0 || yy >= IMG) continue;
            #pragma unroll
            for (int kx = 0; kx < 3; ++kx) {
                int xx = x + kx - 1;
                if (xx < 0 || xx >= IMG) continue;
                s += ip[yy * IMG + xx] * wp[ky * 3 + kx];
            }
        }
    }
    out[idx] = s;  // same [B,C,HW] ordering as thread index
}

// ---------------------------------------------------------------------------
// p=-1 normalization across the 8 heads for fixed (b, c-within-head, pixel):
//   norm = 1 / sum_h (1/|t_h|) ;  out = t / max(norm, eps)  (stored f16)
// ---------------------------------------------------------------------------
__global__ void pm1norm_kernel(const float* __restrict__ t,
                               _Float16* __restrict__ o, int total)
{
    int idx = blockIdx.x * blockDim.x + threadIdx.x;
    if (idx >= total) return;
    int p = idx & (HW - 1);
    int c = (idx >> 14) % HC;
    int b = idx / (HC * HW);
    long long base = (long long)b * ((long long)CDIM * HW) + (long long)c * HW + p;
    float v[NH];
    float sr = 0.0f;
    #pragma unroll
    for (int h = 0; h < NH; ++h) {
        v[h] = t[base + (long long)h * (HC * HW)];
        sr += 1.0f / fabsf(v[h]);
    }
    float norm  = 1.0f / sr;
    float scale = 1.0f / fmaxf(norm, EPSV);
    #pragma unroll
    for (int h = 0; h < NH; ++h)
        o[base + (long long)h * (HC * HW)] = (_Float16)(v[h] * scale);
}

// ---------------------------------------------------------------------------
// Channel attention scores + softmax. One workgroup per (b,h); 8 waves split
// the 16384-pixel reduction; each wave holds a 3x3 grid of 16x16 WMMA tiles
// covering the 48x48 logits; ds_add_f32 cross-wave reduce; softmax in-block.
// ---------------------------------------------------------------------------
__launch_bounds__(256)
__global__ void attn_scores_kernel(const _Float16* __restrict__ qn,
                                   const _Float16* __restrict__ kn,
                                   const float* __restrict__ temp,
                                   float* __restrict__ attn)
{
    __shared__ float S[HC * HC];
    const int bh = blockIdx.x;
    const int b = bh >> 3, h = bh & 7;
    const _Float16* qh  = qn + ((long long)b * CDIM + h * HC) * HW;
    const _Float16* khd = kn + ((long long)b * CDIM + h * HC) * HW;

    for (int i = threadIdx.x; i < HC * HC; i += 256) S[i] = 0.0f;
    __syncthreads();

    const int lane = threadIdx.x & 31;
    const int wave = threadIdx.x >> 5;
    const int kh   = lane >> 4;
    const int l16  = lane & 15;

    v8f acc[3][3] = {};

    const int pBeg = wave * (HW / 8);
    for (int p0 = pBeg; p0 < pBeg + HW / 8; p0 += 32) {
        v16h a[3], bb[3];
        #pragma unroll
        for (int mt = 0; mt < 3; ++mt) {
            const _Float16* ap = qh + (long long)(mt * 16 + l16) * HW + p0 + kh * 16;
            v8h lo = *(const v8h*)ap;          // K = rel {0..7} or {16..23}
            v8h hi = *(const v8h*)(ap + 8);    // K = rel {8..15} or {24..31}
            #pragma unroll
            for (int t = 0; t < 8; ++t) { a[mt][t] = lo[t]; a[mt][8 + t] = hi[t]; }
        }
        #pragma unroll
        for (int nt = 0; nt < 3; ++nt) {
            const _Float16* bp = khd + (long long)(nt * 16 + l16) * HW + p0 + kh * 16;
            bb[nt] = *(const v16h*)bp;         // 16 contiguous K halves, 32B aligned
        }
        #pragma unroll
        for (int mt = 0; mt < 3; ++mt)
            #pragma unroll
            for (int nt = 0; nt < 3; ++nt)
                acc[mt][nt] = __builtin_amdgcn_wmma_f32_16x16x32_f16(
                    false, a[mt], false, bb[nt], (short)0, acc[mt][nt], false, false);
    }

    // cross-wave reduction into LDS (ds_add_f32)
    #pragma unroll
    for (int mt = 0; mt < 3; ++mt)
        #pragma unroll
        for (int nt = 0; nt < 3; ++nt)
            #pragma unroll
            for (int r = 0; r < 8; ++r) {
                int row = mt * 16 + kh * 8 + r;
                int col = nt * 16 + l16;
                atomicAdd(&S[row * HC + col], acc[mt][nt][r]);
            }
    __syncthreads();

    // softmax over d (cols) with per-head temperature
    if (threadIdx.x < HC) {
        int r = threadIdx.x;
        float tp = temp[h];
        float m = -3.4e38f;
        for (int d = 0; d < HC; ++d) m = fmaxf(m, S[r * HC + d] * tp);
        float sum = 0.0f;
        for (int d = 0; d < HC; ++d) sum += expf(S[r * HC + d] * tp - m);
        float inv = 1.0f / sum;
        long long obase = (long long)bh * (HC * HC) + (long long)r * HC;
        for (int d = 0; d < HC; ++d)
            attn[obase + d] = expf(S[r * HC + d] * tp - m) * inv;
    }
}

// ---------------------------------------------------------------------------
extern "C" void kernel_launch(void* const* d_in, const int* in_sizes, int n_in,
                              void* d_out, int out_size, void* d_ws, size_t ws_size,
                              hipStream_t stream)
{
    const float* x    = (const float*)d_in[0];
    const float* ff   = (const float*)d_in[1];
    const float* wq   = (const float*)d_in[2];  // [384,384]
    const float* wkv  = (const float*)d_in[3];  // [768,384]
    const float* wqd  = (const float*)d_in[4];  // [384,9]
    const float* wkvd = (const float*)d_in[5];  // [768,9]
    const float* wout = (const float*)d_in[6];  // [384,384]
    const float* temp = (const float*)d_in[7];  // [8]
    float* out = (float*)d_out;

    // workspace layout (bytes): A(201MB) | Dk(100MB) | Dv(100MB) | Qn(50MB) | Kn(50MB) | Attn
    char* ws = (char*)d_ws;
    const size_t planeQ  = (size_t)NB * CDIM * HW;      // 25,165,824 elems
    const size_t planeKV = 2 * planeQ;
    float*    A    = (float*)ws;
    float*    Dk   = (float*)(ws + planeKV * sizeof(float));
    float*    Dv   = Dk + planeQ;
    _Float16* Qn   = (_Float16*)(Dv + planeQ);
    _Float16* Kn   = Qn + planeQ;
    float*    Attn = (float*)(Kn + planeQ);             // NB*NH*48*48

    const long long s384 = (long long)CDIM * HW;
    const long long s768 = 2 * s384;
    const long long s48  = (long long)HC * HW;
    const int totC  = NB * CDIM * HW;
    const int totN  = NB * HC * HW;

    // 1) q pointwise conv: A = Wq @ ff
    gemm_wmma_kernel<<<dim3(HW / 64, CDIM / 64, NB), 128, 0, stream>>>(
        wq, ff, A, CDIM, CDIM, HW, 0, s384, s384);
    // 2) q depthwise -> Dk
    dwconv3x3_kernel<<<(totC + 255) / 256, 256, 0, stream>>>(
        A, wqd, Dk, CDIM, s384, 0, 0, totC);
    // 3) q normalize -> Qn (f16)
    pm1norm_kernel<<<(totN + 255) / 256, 256, 0, stream>>>(Dk, Qn, totN);
    // 4) kv pointwise conv: A = Wkv @ x
    gemm_wmma_kernel<<<dim3(HW / 64, (2 * CDIM) / 64, NB), 128, 0, stream>>>(
        wkv, x, A, 2 * CDIM, CDIM, HW, 0, s384, s768);
    // 5) k depthwise -> Dk ; 6) v depthwise -> Dv
    dwconv3x3_kernel<<<(totC + 255) / 256, 256, 0, stream>>>(
        A, wkvd, Dk, CDIM, s768, 0, 0, totC);
    dwconv3x3_kernel<<<(totC + 255) / 256, 256, 0, stream>>>(
        A, wkvd, Dv, CDIM, s768, s384, CDIM, totC);
    // 7) k normalize -> Kn (f16)
    pm1norm_kernel<<<(totN + 255) / 256, 256, 0, stream>>>(Dk, Kn, totN);
    // 8) attention scores + softmax -> Attn
    attn_scores_kernel<<<NB * NH, 256, 0, stream>>>(Qn, Kn, temp, Attn);
    // 9) attn @ v -> Dk   (batched over b*h; uniform strides)
    gemm_wmma_kernel<<<dim3(HW / 64, 1, NB * NH), 128, 0, stream>>>(
        Attn, Dv, Dk, HC, HC, HW, (long long)HC * HC, s48, s48);
    // 10) output projection: out = Wout @ Dk
    gemm_wmma_kernel<<<dim3(HW / 64, CDIM / 64, NB), 128, 0, stream>>>(
        wout, Dk, out, CDIM, CDIM, HW, 0, s384, s384);
}
